// SelfAttention_42314017800789
// MI455X (gfx1250) — matrix operand order repaired and checked by reading, not compiled, not run on hardware
//
#include <hip/hip_runtime.h>

typedef __attribute__((ext_vector_type(16))) _Float16 v16h;
typedef __attribute__((ext_vector_type(8)))  _Float16 v8h;
typedef __attribute__((ext_vector_type(8)))  float    v8f;
typedef __attribute__((ext_vector_type(4)))  int      v4i;

#define B_   8
#define C_   512
#define S_   1024
#define NH   8
#define HD   64
#define CPG  64

// workspace layout (in _Float16 elements)
#define OFF_WQ 0
#define N_WQ   (1536 * 512)
#define OFF_WP (OFF_WQ + N_WQ)
#define N_WP   (512 * 512)
#define OFF_H  (OFF_WP + N_WP)
#define N_H    (B_ * S_ * C_)
#define OFF_Q  (OFF_H + N_H)
#define N_QKV  (B_ * NH * S_ * HD)
#define OFF_K  (OFF_Q + N_QKV)
#define OFF_V  (OFF_K + N_QKV)
#define OFF_AO (OFF_V + N_QKV)

#if __has_builtin(__builtin_amdgcn_global_load_async_to_lds_b128)
#define USE_ASYNC_LDS 1
#else
#define USE_ASYNC_LDS 0
#endif

// copy 16 bytes global -> LDS (async-to-LDS on CDNA5 when available)
__device__ inline void stage_b128(const _Float16* g, _Float16* l) {
#if USE_ASYNC_LDS
  __builtin_amdgcn_global_load_async_to_lds_b128(
      (__attribute__((address_space(1))) v4i*)g,
      (__attribute__((address_space(3))) v4i*)l, 0, 0);
#else
  *(v8h*)l = *(const v8h*)g;
#endif
}

__device__ inline void stage_wait() {
#if USE_ASYNC_LDS
#if __has_builtin(__builtin_amdgcn_s_wait_asynccnt)
  __builtin_amdgcn_s_wait_asynccnt(0);
#else
  asm volatile("s_wait_asynccnt 0" ::: "memory");
#endif
#endif
}

// ---- WMMA fragment builders ----
// A: 16xK row-major tile. Lane l holds row l%16;
// 16 contiguous K at k0 + (l/16)*16 + {0..15}.
__device__ inline v16h ld_frag_a(const _Float16* p /* pre-offset row/k */) {
  v8h lo = *(const v8h*)p;
  v8h hi = *(const v8h*)(p + 8);
  v16h r;
#pragma unroll
  for (int i = 0; i < 8; ++i) { r[i] = lo[i]; r[i + 8] = hi[i]; }
  return r;
}
// B: Kx16 tile, K-contiguous per column; lane l holds col l%16,
// 16 contiguous K at k0 + (l/16)*16. p points at the first of 16 halves.
__device__ inline v16h ld_frag16(const _Float16* p) {
  v8h lo = *(const v8h*)p;
  v8h hi = *(const v8h*)(p + 8);
  v16h r;
#pragma unroll
  for (int i = 0; i < 8; ++i) { r[i] = lo[i]; r[i + 8] = hi[i]; }
  return r;
}

// ---- kernel 0: f32 -> f16 weight conversion ----
__global__ void cvt_kernel(const float* __restrict__ src,
                           _Float16* __restrict__ dst, int n) {
  int i = blockIdx.x * blockDim.x + threadIdx.x;
  if (i < n) dst[i] = (_Float16)src[i];
}

// ---- kernel 1: GroupNorm -> h_t[b][s][c] (f16, c-contiguous) ----
__global__ __launch_bounds__(256) void gn_kernel(
    const float* __restrict__ x, const float* __restrict__ w,
    const float* __restrict__ bgn, _Float16* __restrict__ h_t) {
  int bg = blockIdx.x;  // B*GROUPS = 64
  int b = bg >> 3, g = bg & 7;
  const float* base = x + ((size_t)(b * C_ + g * CPG)) * S_;
  const int N = CPG * S_;  // 65536
  float s = 0.f, ss = 0.f;
  for (int i = threadIdx.x; i < N; i += 256) {
    float v = base[i];
    s += v; ss += v * v;
  }
  __shared__ float r1[256], r2[256];
  r1[threadIdx.x] = s; r2[threadIdx.x] = ss;
  __syncthreads();
  for (int off = 128; off > 0; off >>= 1) {
    if (threadIdx.x < (unsigned)off) {
      r1[threadIdx.x] += r1[threadIdx.x + off];
      r2[threadIdx.x] += r2[threadIdx.x + off];
    }
    __syncthreads();
  }
  float mean = r1[0] / (float)N;
  float var  = r2[0] / (float)N - mean * mean;
  float rstd = rsqrtf(var + 1e-5f);
  for (int i = threadIdx.x; i < N; i += 256) {
    int cl = i >> 10, sp = i & 1023;
    int c = g * CPG + cl;
    float v = (base[i] - mean) * rstd * w[c] + bgn[c];
    h_t[((size_t)(b * S_ + sp)) * C_ + c] = (_Float16)v;
  }
}

// ---- kernel 2: QKV GEMM. Block = 128(o) x 64(s); B panel staged in LDS ----
__global__ __launch_bounds__(256) void qkv_kernel(
    const _Float16* __restrict__ Wq, const float* __restrict__ bias,
    const _Float16* __restrict__ h_t, _Float16* __restrict__ qt,
    _Float16* __restrict__ kt, _Float16* __restrict__ vt) {
  __shared__ _Float16 b_s[64 * 512];  // 64 KB: [s_local][c]
  int bt = blockIdx.x;                 // B * 12 * 16 = 1536
  int b = bt / (12 * 16);
  int r = bt % (12 * 16);
  int obt = r >> 4, st = r & 15;
  int o_base = obt * 128, s0 = st * 64;
  const _Float16* hb = h_t + ((size_t)(b * S_ + s0)) * C_;  // contiguous panel
#pragma unroll
  for (int i = 0; i < 16; ++i) {
    int g = threadIdx.x + i * 256;  // 4096 groups of 8 halves
    stage_b128(hb + (size_t)g * 8, &b_s[g * 8]);
  }
  stage_wait();
  __syncthreads();

  int wave = threadIdx.x >> 5, lane = threadIdx.x & 31;
  int nl = lane & 15, hf = lane >> 4;
  int o0 = o_base + wave * 16;
  const _Float16* arow = Wq + ((size_t)(o0 + nl)) * C_ + hf * 16;
  v8f acc[4] = {};
  for (int k0 = 0; k0 < C_; k0 += 32) {
    v16h a = ld_frag_a(arow + k0);
#pragma unroll
    for (int j = 0; j < 4; ++j) {
      v16h bf = ld_frag16(&b_s[(j * 16 + nl) * 512 + k0 + hf * 16]);
      acc[j] = __builtin_amdgcn_wmma_f32_16x16x32_f16(false, a, false, bf,
                                                      (short)0, acc[j], false, false);
    }
  }
#pragma unroll
  for (int j = 0; j < 4; ++j) {
    int s = s0 + j * 16 + nl;
#pragma unroll
    for (int i = 0; i < 8; ++i) {
      int o = o0 + i + hf * 8;
      float v = acc[j][i] + bias[o];
      if (o < 512) {           // q, pre-scaled by hd^-0.5
        int hn = o >> 6, c = o & 63;
        qt[(((size_t)(b * NH + hn)) * S_ + s) * HD + c] = (_Float16)(v * 0.125f);
      } else if (o < 1024) {   // k
        int o2 = o - 512; int hn = o2 >> 6, c = o2 & 63;
        kt[(((size_t)(b * NH + hn)) * S_ + s) * HD + c] = (_Float16)v;
      } else {                 // v : [b][n][c][s]
        int o2 = o - 1024; int hn = o2 >> 6, c = o2 & 63;
        vt[(((size_t)(b * NH + hn)) * HD + c) * S_ + s] = (_Float16)v;
      }
    }
  }
}

// ---- kernel 3: flash attention; K/V chunks staged in LDS per block ----
__global__ __launch_bounds__(256) void attn_kernel(
    const _Float16* __restrict__ qt, const _Float16* __restrict__ kt,
    const _Float16* __restrict__ vt, _Float16* __restrict__ ao) {
  __shared__ _Float16 k_s[128 * 64];       // 16 KB: [f_local][c]
  __shared__ _Float16 v_s[64 * 128];       // 16 KB: [c][f_local]
  __shared__ _Float16 p_s[8][16 * 128];    // 32 KB: per-wave P staging
  int wave = threadIdx.x >> 5, lane = threadIdx.x & 31;
  int nl = lane & 15, hf = lane >> 4;
  int blk = blockIdx.x;  // B*NH*8 = 512
  int b = blk / (NH * 8);
  int rr = blk % (NH * 8);
  int hn = rr >> 3, dt = rr & 7;
  int d0 = dt * 128 + wave * 16;
  const _Float16* qb = qt + ((size_t)(b * NH + hn)) * S_ * HD;
  const _Float16* kb = kt + ((size_t)(b * NH + hn)) * S_ * HD;
  const _Float16* vb = vt + ((size_t)(b * NH + hn)) * HD * S_;
  const _Float16* qrow = qb + ((size_t)(d0 + nl)) * HD + hf * 16;
  v16h qa0 = ld_frag_a(qrow);
  v16h qa1 = ld_frag_a(qrow + 32);
  float rowmax[8], rowsum[8];
#pragma unroll
  for (int i = 0; i < 8; ++i) { rowmax[i] = -3.0e38f; rowsum[i] = 0.f; }
  v8f accO[4] = {};

  for (int f0 = 0; f0 < S_; f0 += 128) {
    // stage K chunk: rows f0..f0+127 of [s][64] -> contiguous 8192 halves
    const _Float16* kg = kb + (size_t)f0 * HD;
#pragma unroll
    for (int i = 0; i < 4; ++i) {
      int g = threadIdx.x + i * 256;     // 1024 groups of 8 halves
      stage_b128(kg + g * 8, &k_s[g * 8]);
    }
    // stage V chunk: 64 rows x 128 cols (row stride S_)
#pragma unroll
    for (int i = 0; i < 4; ++i) {
      int g = threadIdx.x + i * 256;
      int c = g >> 4, j = (g & 15) * 8;
      stage_b128(vb + (size_t)c * S_ + f0 + j, &v_s[c * 128 + j]);
    }
    stage_wait();
    __syncthreads();

    v8f Sa[8];
#pragma unroll
    for (int ct = 0; ct < 8; ++ct) {
      v8f sacc = {};
      v16h bk0 = ld_frag16(&k_s[(ct * 16 + nl) * 64 + hf * 16]);
      sacc = __builtin_amdgcn_wmma_f32_16x16x32_f16(false, qa0, false, bk0,
                                                    (short)0, sacc, false, false);
      v16h bk1 = ld_frag16(&k_s[(ct * 16 + nl) * 64 + 32 + hf * 16]);
      sacc = __builtin_amdgcn_wmma_f32_16x16x32_f16(false, qa1, false, bk1,
                                                    (short)0, sacc, false, false);
      Sa[ct] = sacc;
    }
    // online softmax: row = i + 8*hf, columns spread over the 16-lane half-wave
#pragma unroll
    for (int i = 0; i < 8; ++i) {
      float m = Sa[0][i];
#pragma unroll
      for (int ct = 1; ct < 8; ++ct) m = fmaxf(m, Sa[ct][i]);
      m = fmaxf(m, __shfl_xor(m, 1, 16));
      m = fmaxf(m, __shfl_xor(m, 2, 16));
      m = fmaxf(m, __shfl_xor(m, 4, 16));
      m = fmaxf(m, __shfl_xor(m, 8, 16));
      float nm = fmaxf(rowmax[i], m);
      float sc = __expf(rowmax[i] - nm);
      rowmax[i] = nm;
      rowsum[i] *= sc;
#pragma unroll
      for (int j = 0; j < 4; ++j) accO[j][i] *= sc;
      float ps = 0.f;
#pragma unroll
      for (int ct = 0; ct < 8; ++ct) {
        float p = __expf(Sa[ct][i] - nm);
        ps += p;
        p_s[wave][(i + 8 * hf) * 128 + ct * 16 + nl] = (_Float16)p;
      }
      ps += __shfl_xor(ps, 1, 16);
      ps += __shfl_xor(ps, 2, 16);
      ps += __shfl_xor(ps, 4, 16);
      ps += __shfl_xor(ps, 8, 16);
      rowsum[i] += ps;
    }
    // P(16x128) x V^T(128x64) accumulate
#pragma unroll
    for (int fk = 0; fk < 4; ++fk) {
      v16h pa = ld_frag_a(&p_s[wave][nl * 128 + fk * 32 + hf * 16]);
#pragma unroll
      for (int ct2 = 0; ct2 < 4; ++ct2) {
        v16h bv = ld_frag16(&v_s[(ct2 * 16 + nl) * 128 + fk * 32 + hf * 16]);
        accO[ct2] = __builtin_amdgcn_wmma_f32_16x16x32_f16(false, pa, false, bv,
                                                           (short)0, accO[ct2], false, false);
      }
    }
    __syncthreads();  // staging buffers reused next chunk
  }
  // epilogue: divide by rowsum, write ao[b][s][channel] (c-contiguous for proj B)
#pragma unroll
  for (int i = 0; i < 8; ++i) {
    float inv = 1.f / rowsum[i];
    int s = d0 + i + 8 * hf;
#pragma unroll
    for (int ct2 = 0; ct2 < 4; ++ct2) {
      int cch = hn * HD + ct2 * 16 + nl;
      ao[((size_t)(b * S_ + s)) * C_ + cch] = (_Float16)(accO[ct2][i] * inv);
    }
  }
}

// ---- kernel 4: proj GEMM + bias + residual. Block = 128(o) x 64(s) ----
__global__ __launch_bounds__(256) void proj_kernel(
    const _Float16* __restrict__ Wp, const float* __restrict__ bias,
    const _Float16* __restrict__ ao, const float* __restrict__ x,
    float* __restrict__ out) {
  __shared__ _Float16 b_s[64 * 512];  // 64 KB
  int bt = blockIdx.x;                 // B * 4 * 16 = 512
  int b = bt / (4 * 16);
  int r = bt % (4 * 16);
  int obt = r >> 4, st = r & 15;
  int o_base = obt * 128, s0 = st * 64;
  const _Float16* ab = ao + ((size_t)(b * S_ + s0)) * C_;
#pragma unroll
  for (int i = 0; i < 16; ++i) {
    int g = threadIdx.x + i * 256;
    stage_b128(ab + (size_t)g * 8, &b_s[g * 8]);
  }
  stage_wait();
  __syncthreads();

  int wave = threadIdx.x >> 5, lane = threadIdx.x & 31;
  int nl = lane & 15, hf = lane >> 4;
  int o0 = o_base + wave * 16;
  const _Float16* arow = Wp + ((size_t)(o0 + nl)) * C_ + hf * 16;
  v8f acc[4] = {};
  for (int k0 = 0; k0 < C_; k0 += 32) {
    v16h a = ld_frag_a(arow + k0);
#pragma unroll
    for (int j = 0; j < 4; ++j) {
      v16h bf = ld_frag16(&b_s[(j * 16 + nl) * 512 + k0 + hf * 16]);
      acc[j] = __builtin_amdgcn_wmma_f32_16x16x32_f16(false, a, false, bf,
                                                      (short)0, acc[j], false, false);
    }
  }
#pragma unroll
  for (int j = 0; j < 4; ++j) {
    int s = s0 + j * 16 + nl;
#pragma unroll
    for (int i = 0; i < 8; ++i) {
      int o = o0 + i + 8 * hf;
      size_t idx = ((size_t)(b * C_ + o)) * S_ + s;
      out[idx] = x[idx] + bias[o] + acc[j][i];
    }
  }
}

extern "C" void kernel_launch(void* const* d_in, const int* in_sizes, int n_in,
                              void* d_out, int out_size, void* d_ws, size_t ws_size,
                              hipStream_t stream) {
  const float* x     = (const float*)d_in[0];
  const float* gnw   = (const float*)d_in[1];
  const float* gnb   = (const float*)d_in[2];
  const float* qkvw  = (const float*)d_in[3];
  const float* qkvb  = (const float*)d_in[4];
  const float* projw = (const float*)d_in[5];
  const float* projb = (const float*)d_in[6];

  _Float16* ws  = (_Float16*)d_ws;
  _Float16* Wq  = ws + OFF_WQ;
  _Float16* Wp  = ws + OFF_WP;
  _Float16* h_t = ws + OFF_H;
  _Float16* qt  = ws + OFF_Q;
  _Float16* kt  = ws + OFF_K;
  _Float16* vt  = ws + OFF_V;
  _Float16* ao  = ws + OFF_AO;

  cvt_kernel<<<(N_WQ + 255) / 256, 256, 0, stream>>>(qkvw, Wq, N_WQ);
  cvt_kernel<<<(N_WP + 255) / 256, 256, 0, stream>>>(projw, Wp, N_WP);
  gn_kernel<<<B_ * 8, 256, 0, stream>>>(x, gnw, gnb, h_t);
  qkv_kernel<<<B_ * 12 * 16, 256, 0, stream>>>(Wq, qkvb, h_t, qt, kt, vt);
  attn_kernel<<<B_ * NH * 8, 256, 0, stream>>>(qt, kt, vt, ao);
  proj_kernel<<<B_ * 4 * 16, 256, 0, stream>>>(Wp, projb, ao, x, (float*)d_out);
}
